// SimpleVQ_52123723105119
// MI455X (gfx1250) — matrix operand order repaired and checked by reading, not compiled, not run on hardware
//
#include <hip/hip_runtime.h>

// ---------------------------------------------------------------------------
// CDNA5 (gfx1250) VQ: scores via split-bf16 WMMA, fused argmin epilogue.
// Round 4: amdgpu_waves_per_eu(1,2) to free the register allocator and
// eliminate the residual b128 scratch spills around tile staging.
// ---------------------------------------------------------------------------

typedef __attribute__((ext_vector_type(16))) __bf16 v16bf;
typedef __attribute__((ext_vector_type(8)))  float  v8f;

#define VQ_B 32768
#define VQ_D 512
#define VQ_K 8192
#define SLICES 16          // D / 32 (WMMA K-depth = 32 for bf16)
#define ROWS_PER_BLOCK 128 // 8 waves * 16 rows
#define TILE_K 16          // codebook rows per LDS tile

__device__ __forceinline__ unsigned short f2bf(float f) {
    // round-to-nearest-even fp32 -> bf16
    unsigned u = __builtin_bit_cast(unsigned, f);
    unsigned r = u + 0x7FFFu + ((u >> 16) & 1u);
    return (unsigned short)(r >> 16);
}
__device__ __forceinline__ float bf2f(unsigned short h) {
    return __builtin_bit_cast(float, (unsigned)h << 16);
}

union FragU {
    v16bf v;
    unsigned short u[16];
};

// ---------------------------------------------------------------------------
// Kernel 1: codebook -> bf16 hi/lo split, per-row squared norm, zero loss slot
// ---------------------------------------------------------------------------
__global__ void vq_prep(const float* __restrict__ cb,
                        unsigned short* __restrict__ c_hi,
                        unsigned short* __restrict__ c_lo,
                        float* __restrict__ c_sq,
                        float* __restrict__ loss_slot) {
    __shared__ float red[256];
    const int k = blockIdx.x;
    const int tid = threadIdx.x;
    float acc = 0.f;
    for (int d = tid; d < VQ_D; d += 256) {
        float f = cb[(size_t)k * VQ_D + d];
        unsigned short hi = f2bf(f);
        unsigned short lo = f2bf(f - bf2f(hi));
        c_hi[(size_t)k * VQ_D + d] = hi;
        c_lo[(size_t)k * VQ_D + d] = lo;
        acc += f * f;
    }
    red[tid] = acc;
    __syncthreads();
    for (int s = 128; s > 0; s >>= 1) {
        if (tid < s) red[tid] += red[tid + s];
        __syncthreads();
    }
    if (tid == 0) c_sq[k] = red[0];
    if (k == 0 && tid == 0) *loss_slot = 0.f;  // must re-zero every call
}

// ---------------------------------------------------------------------------
// Kernel 2: fused score GEMM + argmin.
// score(b,k) = ||c_k||^2 - 2 * dot(x_b, c_k);  dot via hi*hi + hi*lo + lo*hi.
// Pipeline: tile t+1 global loads in flight while tile t computes; LDS
// double-buffered so one barrier per iteration suffices.
// amdgpu_waves_per_eu(1,2): allow the RA up to a full-file allocation (no
// scratch spills for the ~360 live VGPRs) while targeting 2 waves/SIMD.
// ---------------------------------------------------------------------------
__global__ __launch_bounds__(256)
__attribute__((amdgpu_waves_per_eu(1, 2)))
void vq_argmin_kernel(
    const float* __restrict__ x,
    const unsigned short* __restrict__ c_hi,
    const unsigned short* __restrict__ c_lo,
    const float* __restrict__ c_sq,
    int* __restrict__ amin) {
    // 2 buffers * (hi + lo) * 16 rows * 512 dims * 2B = 64 KB LDS
    __shared__ __align__(32) unsigned short sh_hi[2 * TILE_K * VQ_D];
    __shared__ __align__(32) unsigned short sh_lo[2 * TILE_K * VQ_D];

    const int tid = threadIdx.x;
    const int wave = tid >> 5;
    const int lane = tid & 31;
    const int laneHi = lane >> 4;  // half-wave select
    const int lr = lane & 15;
    const int row = blockIdx.x * ROWS_PER_BLOCK + wave * 16 + lr;

    // --- Load this wave's 16 x-rows as resident A fragments (hi + lo) -----
    // A k order used here (contiguous, same as B): lanes 0-15 hold M=lane, elems K 0..15 contiguous;
    // lanes 16-31 hold M=lane-16, elems K 16..31 contiguous.
    FragU a_hi[SLICES], a_lo[SLICES];
#pragma unroll
    for (int s = 0; s < SLICES; ++s) {
        const float* px = x + (size_t)row * VQ_D + s * 32 + laneHi * 16;
        float f[16];
        *(float4*)&f[0]  = *(const float4*)(px + 0);
        *(float4*)&f[4]  = *(const float4*)(px + 4);
        *(float4*)&f[8]  = *(const float4*)(px + 8);
        *(float4*)&f[12] = *(const float4*)(px + 12);
#pragma unroll
        for (int j = 0; j < 16; ++j) {
            unsigned short h = f2bf(f[j]);
            a_hi[s].u[j] = h;
            a_lo[s].u[j] = f2bf(f[j] - bf2f(h));
        }
    }

    float minv[8];
    int   mini[8];
#pragma unroll
    for (int i = 0; i < 8; ++i) { minv[i] = __builtin_inff(); mini[i] = 0; }

    // Per-thread staging registers: 64B of hi + 64B of lo per tile.
    int4 rh[4], rl[4];
    auto load_tile_regs = [&](int k0) {
        const size_t src = (size_t)k0 * VQ_D + tid * 32;
        const int4* shi = (const int4*)(c_hi + src);
        const int4* slo = (const int4*)(c_lo + src);
#pragma unroll
        for (int q = 0; q < 4; ++q) { rh[q] = shi[q]; rl[q] = slo[q]; }
    };

    load_tile_regs(0);  // prologue: tile 0 in flight

    for (int k0 = 0; k0 < VQ_K; k0 += TILE_K) {
        const int buf = (k0 >> 4) & 1;
        // Drain staged registers into this iteration's LDS buffer.
        {
            int4* dh = (int4*)(sh_hi + buf * TILE_K * VQ_D + tid * 32);
            int4* dl = (int4*)(sh_lo + buf * TILE_K * VQ_D + tid * 32);
#pragma unroll
            for (int q = 0; q < 4; ++q) { dh[q] = rh[q]; dl[q] = rl[q]; }
        }
        __syncthreads();  // stores visible; prior buffer's readers done 2 barriers ago

        // Kick off next tile's global loads; they fly during the WMMA block.
        if (k0 + TILE_K < VQ_K) load_tile_regs(k0 + TILE_K);

        // B fragment (bf16 32x16): lanes 0-15: N=lane, K=0..15 contiguous;
        // lanes 16-31: N=lane-16, K=16..31 contiguous  -> 32B per-lane reads.
        const unsigned short* bhi = sh_hi + buf * TILE_K * VQ_D + lr * VQ_D + laneHi * 16;
        const unsigned short* blo = sh_lo + buf * TILE_K * VQ_D + lr * VQ_D + laneHi * 16;
        v8f c = {};
#pragma unroll
        for (int s = 0; s < SLICES; ++s) {
            const v16bf bh = *(const v16bf*)(bhi + s * 32);
            const v16bf bl = *(const v16bf*)(blo + s * 32);
            c = __builtin_amdgcn_wmma_f32_16x16x32_bf16(
                    false, a_hi[s].v, false, bh, (short)0, c, false, false);
            c = __builtin_amdgcn_wmma_f32_16x16x32_bf16(
                    false, a_hi[s].v, false, bl, (short)0, c, false, false);
            c = __builtin_amdgcn_wmma_f32_16x16x32_bf16(
                    false, a_lo[s].v, false, bh, (short)0, c, false, false);
        }

        // Epilogue: C layout (lane, vgpr i) -> row = i + 8*laneHi, col = lr.
        const float csq = c_sq[k0 + lr];
        const int kidx = k0 + lr;
#pragma unroll
        for (int i = 0; i < 8; ++i) {
            float sc = csq - 2.0f * c[i];
            if (sc < minv[i]) { minv[i] = sc; mini[i] = kidx; }
        }
    }

    // Reduce across the 16 lanes (column classes) sharing each row.
#pragma unroll
    for (int off = 1; off < 16; off <<= 1) {
#pragma unroll
        for (int i = 0; i < 8; ++i) {
            float ov = __shfl_xor(minv[i], off, 32);
            int   oi = __shfl_xor(mini[i], off, 32);
            if (ov < minv[i] || (ov == minv[i] && oi < mini[i])) {
                minv[i] = ov; mini[i] = oi;
            }
        }
    }
    if (lr == 0) {
#pragma unroll
        for (int i = 0; i < 8; ++i)
            amin[blockIdx.x * ROWS_PER_BLOCK + wave * 16 + laneHi * 8 + i] = mini[i];
    }
}

// ---------------------------------------------------------------------------
// Kernel 3: gather winners (fp32), write indices, accumulate commitment loss.
// quantized_ste forward value == codebook[idx] exactly.
// ---------------------------------------------------------------------------
__global__ void vq_gather(const float* __restrict__ x,
                          const float* __restrict__ cb,
                          const int* __restrict__ amin,
                          float* __restrict__ qout,
                          float* __restrict__ idxout,
                          float* __restrict__ loss) {
    __shared__ float red[256];
    const int b = blockIdx.x;
    const int tid = threadIdx.x;
    const int idx = amin[b];
    const float* crow = cb + (size_t)idx * VQ_D;
    const float* xrow = x + (size_t)b * VQ_D;
    float* orow = qout + (size_t)b * VQ_D;
    float acc = 0.f;
    for (int d = tid; d < VQ_D; d += 256) {
        float q = crow[d];
        float df = xrow[d] - q;
        orow[d] = q;
        acc += df * df;
    }
    red[tid] = acc;
    __syncthreads();
    for (int s = 128; s > 0; s >>= 1) {
        if (tid < s) red[tid] += red[tid + s];
        __syncthreads();
    }
    if (tid == 0) {
        atomicAdd(loss, red[0] * (1.0f / ((float)VQ_B * (float)VQ_D)));
        idxout[b] = (float)idx;
    }
}

// ---------------------------------------------------------------------------
extern "C" void kernel_launch(void* const* d_in, const int* in_sizes, int n_in,
                              void* d_out, int out_size, void* d_ws, size_t ws_size,
                              hipStream_t stream) {
    const float* x  = (const float*)d_in[0];   // (B, D) fp32
    const float* cb = (const float*)d_in[1];   // (K, D) fp32

    // Workspace layout (~16.2 MB total):
    char* ws = (char*)d_ws;
    unsigned short* c_hi = (unsigned short*)ws;                              // 8 MB
    unsigned short* c_lo = (unsigned short*)(ws + (size_t)VQ_K * VQ_D * 2);  // 8 MB
    float* c_sq = (float*)(ws + (size_t)VQ_K * VQ_D * 4);                    // 32 KB
    int*   amin = (int*)(ws + (size_t)VQ_K * VQ_D * 4 + (size_t)VQ_K * 4);   // 128 KB

    float* qout   = (float*)d_out;                   // B*D quantized (== gather)
    float* idxout = qout + (size_t)VQ_B * VQ_D;      // B indices (as float)
    float* loss   = idxout + VQ_B;                   // 1 commitment loss

    vq_prep<<<VQ_K, 256, 0, stream>>>(cb, c_hi, c_lo, c_sq, loss);
    vq_argmin_kernel<<<VQ_B / ROWS_PER_BLOCK, 256, 0, stream>>>(x, c_hi, c_lo, c_sq, amin);
    vq_gather<<<VQ_B, 256, 0, stream>>>(x, cb, amin, qout, idxout, loss);
}
